// Semantic_ConBlock_47682726920293
// MI455X (gfx1250) — matrix operand order repaired and checked by reading, not compiled, not run on hardware
//
#include <hip/hip_runtime.h>

typedef _Float16 f16;
typedef __attribute__((ext_vector_type(16))) _Float16 v16h;
typedef __attribute__((ext_vector_type(8)))  _Float16 v8h;
typedef __attribute__((ext_vector_type(8)))  float    v8f;

#define B_    64
#define C_    512
#define L_    192
#define J_    22
#define LT_   8
#define TD_   768
#define S_    200
#define SPAD_ 208   // padded sequence rows for xa_f16
#define KPAD_ 224   // padded K for attention PV gemm (7 * 32)
#define H_    8
#define D_    64
#define NEG_  (-1000000000.0f)

__device__ __forceinline__ v8f v8f_zero() {
  v8f z = {0.f,0.f,0.f,0.f,0.f,0.f,0.f,0.f};
  return z;
}
__device__ __forceinline__ float silu_f(float x) { return x / (1.0f + expf(-x)); }

// A-fragment style: halves K=ks..ks+7 and K=ks+16..ks+23 (two 16B chunks)
__device__ __forceinline__ v16h ld_frag_split(const f16* p0, const f16* p1) {
  v8h lo = *(const v8h*)p0;
  v8h hi = *(const v8h*)p1;
  v16h r;
  #pragma unroll
  for (int i = 0; i < 8; ++i) { r[i] = lo[i]; r[i+8] = hi[i]; }
  return r;
}
// B-fragment style: halves K=ks..ks+7 and K=ks+16..ks+23 (two 16B chunks)
__device__ __forceinline__ v16h ld_frag16(const f16* p) {
  return ld_frag_split(p, p + 16);
}
__device__ __forceinline__ v8f wmma16(v16h a, v16h b, v8f c) {
  return __builtin_amdgcn_wmma_f32_16x16x32_f16(false, a, false, b, (short)0, c,
                                                false, false);
}

// LDS byte offset of a generic pointer that points into __shared__ memory
__device__ __forceinline__ unsigned lds_addr_of(const void* p) {
  return (unsigned)(unsigned long long)
         (const __attribute__((address_space(3))) char*)p;
}
// async global -> LDS, 16 bytes per lane, tracked by ASYNCcnt
__device__ __forceinline__ void async_g2l_b128(unsigned lds_off, const void* gptr) {
  asm volatile("global_load_async_to_lds_b128 %0, %1, off"
               :: "v"(lds_off), "v"((unsigned long long)(size_t)gptr)
               : "memory");
}

// ---------------------------------------------------------------- converts
__global__ void cvt_f32_to_f16_k(const float* __restrict__ in, f16* __restrict__ out, int n) {
  int i = blockIdx.x * blockDim.x + threadIdx.x;
  int stride = gridDim.x * blockDim.x;
  for (; i < n; i += stride) out[i] = (f16)in[i];
}

// ------------------------------------------------- depthwise (J=22) + SiLU
__global__ void dwconv_silu_k(const float* __restrict__ x, const float* __restrict__ w,
                              const float* __restrict__ bias, f16* __restrict__ out) {
  int b = blockIdx.z;
  int c = blockIdx.y * 8 + (threadIdx.x >> 5);
  int l = blockIdx.x * 32 + (threadIdx.x & 31);
  const float* xp = x + (((long)b * C_ + c) * L_ + l) * J_;
  const float* wrow = w + c * J_;
  float s = bias[c];
  #pragma unroll
  for (int j = 0; j < J_; ++j) s += xp[j] * wrow[j];
  out[((long)b * L_ + l) * C_ + c] = (f16)silu_f(s);
}

// ------------------------------------------------------------ generic GEMM
// out[b, m, n] = sum_k A[b, m, k] * W[n, k] + bias[n]   (N mult of 64, K mult of 32)
// 128 threads = 4 waves; wave w owns rows [mbase+16w, +16), all 64 cols.
// B tile (64 cols x 32 k, 4KB) double-buffered in LDS via async b128 loads;
// A fragments double-buffered in registers; last K-step peeled so the
// steady-state loop body is branch-free.
__global__ void gemm_wmma_k(const f16* __restrict__ A, long bsA, int lda,
                            const f16* __restrict__ W, int ldb, int K,
                            const float* __restrict__ bias,
                            float* __restrict__ out32, long bs32,
                            f16* __restrict__ out16, long bs16,
                            int ldc, int M, int act) {
  __shared__ __align__(16) f16 sB[2][64 * 32];   // [buf][ncol_local*32 + kk]

  int b     = blockIdx.z;
  int nbase = blockIdx.x * 64;
  int mbase = blockIdx.y * 64;
  int tid   = threadIdx.x;
  int wave  = tid >> 5;
  int lane  = tid & 31;
  int mrow0 = mbase + wave * 16;

  const f16* Ab = A + (long)b * bsA;
  int arow = mrow0 + (lane & 15);
  if (arow > M - 1) arow = M - 1;           // clamp (stores are predicated)
  const f16* Arow = Ab + (long)arow * lda;

  const int aks = (lane < 16) ? 0 : 8;      // A: K-halves ks..+7 and ks+16..+23
  const int bk0 = (lane < 16) ? 0 : 8;      // B: K-halves ks..+7 and ks+16..+23
  const int ln16 = lane & 15;

  // each thread pushes two 16B chunks of the B tile per K-step
  int ch0 = tid, ch1 = tid + 128;           // chunk ids 0..255
  int ncl0 = ch0 >> 2, ko0 = (ch0 & 3) * 8;
  int ncl1 = ch1 >> 2, ko1 = (ch1 & 3) * 8;
  const f16* Wg0 = W + (long)(nbase + ncl0) * ldb + ko0;
  const f16* Wg1 = W + (long)(nbase + ncl1) * ldb + ko1;
  unsigned lB[2][2];
  lB[0][0] = lds_addr_of(&sB[0][ncl0 * 32 + ko0]);
  lB[0][1] = lds_addr_of(&sB[0][ncl1 * 32 + ko1]);
  lB[1][0] = lds_addr_of(&sB[1][ncl0 * 32 + ko0]);
  lB[1][1] = lds_addr_of(&sB[1][ncl1 * 32 + ko1]);

  v8f acc[4];
  #pragma unroll
  for (int t = 0; t < 4; ++t) acc[t] = v8f_zero();

  // prologue: B tile for k=0 into buf 0; A fragment for k=0
  async_g2l_b128(lB[0][0], Wg0);
  async_g2l_b128(lB[0][1], Wg1);
  v16h a_cur = ld_frag_split(Arow + aks, Arow + aks + 16);

  int k = 0;
  // steady state: branch-free body, always prefetches the next buffer
  for (; k + 32 < K; k += 32) {
    int buf = (k >> 5) & 1;
    async_g2l_b128(lB[buf ^ 1][0], Wg0 + (k + 32));
    async_g2l_b128(lB[buf ^ 1][1], Wg1 + (k + 32));
    asm volatile("s_wait_asynccnt 0x2" ::: "memory");   // current buf landed
    __syncthreads();

    v16h a_next = ld_frag_split(Arow + (k + 32) + aks, Arow + (k + 32) + aks + 16);

    const f16* sBb = &sB[buf][0];
    // load all four B fragments first, then issue the 4 WMMAs back-to-back
    v16h bf0 = ld_frag16(sBb + (0 * 16 + ln16) * 32 + bk0);
    v16h bf1 = ld_frag16(sBb + (1 * 16 + ln16) * 32 + bk0);
    v16h bf2 = ld_frag16(sBb + (2 * 16 + ln16) * 32 + bk0);
    v16h bf3 = ld_frag16(sBb + (3 * 16 + ln16) * 32 + bk0);
    acc[0] = wmma16(a_cur, bf0, acc[0]);
    acc[1] = wmma16(a_cur, bf1, acc[1]);
    acc[2] = wmma16(a_cur, bf2, acc[2]);
    acc[3] = wmma16(a_cur, bf3, acc[3]);

    a_cur = a_next;
    __syncthreads();                         // reads done before buf is rewritten
  }
  { // peeled final K-step: nothing left to prefetch
    int buf = (k >> 5) & 1;
    asm volatile("s_wait_asynccnt 0x0" ::: "memory");
    __syncthreads();
    const f16* sBb = &sB[buf][0];
    v16h bf0 = ld_frag16(sBb + (0 * 16 + ln16) * 32 + bk0);
    v16h bf1 = ld_frag16(sBb + (1 * 16 + ln16) * 32 + bk0);
    v16h bf2 = ld_frag16(sBb + (2 * 16 + ln16) * 32 + bk0);
    v16h bf3 = ld_frag16(sBb + (3 * 16 + ln16) * 32 + bk0);
    acc[0] = wmma16(a_cur, bf0, acc[0]);
    acc[1] = wmma16(a_cur, bf1, acc[1]);
    acc[2] = wmma16(a_cur, bf2, acc[2]);
    acc[3] = wmma16(a_cur, bf3, acc[3]);
  }

  int rowtop = (lane >> 4) << 3;            // 0 or 8
  #pragma unroll
  for (int t = 0; t < 4; ++t) {
    int ncol = nbase + t * 16 + ln16;
    float bv = bias[ncol];
    #pragma unroll
    for (int r = 0; r < 8; ++r) {
      int row = mrow0 + rowtop + r;
      if (row < M) {
        float v = acc[t][r] + bv;
        if (act) v = silu_f(v);
        if (out32) out32[(long)b * bs32 + (long)row * ldc + ncol] = v;
        if (out16) out16[(long)b * bs16 + (long)row * ldc + ncol] = (f16)v;
      }
    }
  }
}

// -------------------------------------------------------- fused attention
__global__ void attn_k(const f16* __restrict__ xa16, const float* __restrict__ rpe,
                       const unsigned char* __restrict__ tm1,
                       const unsigned char* __restrict__ xm,
                       f16* __restrict__ out16) {
  __shared__ __align__(16) f16   sh_qT[D_ * KPAD_];    // qT[d][j], zero-padded j>=200
  __shared__ __align__(16) float sh_dots[16 * KPAD_];
  __shared__ __align__(16) f16   sh_p[16 * KPAD_];
  __shared__ float sh_maskb[KPAD_];
  __shared__ float sh_red[16 * 8];
  __shared__ float sh_rowmax[16];
  __shared__ float sh_rowsum[16];

  int b = blockIdx.z, h = blockIdx.y, mtile = blockIdx.x;
  int tid = threadIdx.x;
  const f16* xab = xa16 + (long)b * SPAD_ * C_ + h * D_;

  for (int t = tid; t < KPAD_ * D_; t += 128) {
    int j = t >> 6, d = t & 63;
    f16 v = (f16)0.f;
    if (j < S_) v = xab[(long)j * C_ + d];
    sh_qT[d * KPAD_ + j] = v;
  }
  for (int t = tid; t < KPAD_; t += 128) {
    bool ok = false;
    if (t < LT_)      ok = tm1[b * LT_ + t] != 0;
    else if (t < S_)  ok = xm[b * L_ + (t - LT_)] != 0;
    sh_maskb[t] = ok ? 0.0f : NEG_;
  }
  __syncthreads();

  int wave = tid >> 5, lane = tid & 31;
  const int aks = (lane < 16) ? 0 : 8;
  const int bk0 = (lane < 16) ? 0 : 8;
  int rowtop = (lane >> 4) << 3;

  { // phase 1: dots = q . qT (K = D = 64 -> 2 wmma per tile)
    int arow = mtile * 16 + (lane & 15);       // < 208: inside padded xa16
    const f16* Ap = xab + (long)arow * C_;
    v16h a0 = ld_frag_split(Ap + aks,      Ap + aks + 16);
    v16h a1 = ld_frag_split(Ap + 32 + aks, Ap + 32 + aks + 16);
    for (int nt = wave; nt < 13; nt += 4) {    // uniform per wave: EXEC stays full
      int j = nt * 16 + (lane & 15);
      const f16* Bp = xab + (long)j * C_;
      v16h b0 = ld_frag16(Bp + bk0);
      v16h b1 = ld_frag16(Bp + 32 + bk0);
      v8f c = v8f_zero();
      c = wmma16(a0, b0, c);
      c = wmma16(a1, b1, c);
      int col = nt * 16 + (lane & 15);
      #pragma unroll
      for (int r = 0; r < 8; ++r) {
        int i = mtile * 16 + rowtop + r;
        float v = NEG_;
        if (i < S_ && col < S_)
          v = c[r] * 0.125f + rpe[h * 399 + (col - i + 199)] + sh_maskb[col];
        sh_dots[(rowtop + r) * KPAD_ + col] = v;
      }
    }
  }
  __syncthreads();

  { // softmax over 224 cols, 8 threads per row
    int row = tid >> 3, sub = tid & 7;
    float m = NEG_;
    for (int j = sub; j < KPAD_; j += 8) m = fmaxf(m, sh_dots[row * KPAD_ + j]);
    sh_red[row * 8 + sub] = m;
    __syncthreads();
    if (sub == 0) {
      float mm = sh_red[row * 8];
      for (int k2 = 1; k2 < 8; ++k2) mm = fmaxf(mm, sh_red[row * 8 + k2]);
      sh_rowmax[row] = mm;
    }
    __syncthreads();
    float mm = sh_rowmax[row];
    float ssum = 0.f;
    for (int j = sub; j < KPAD_; j += 8) {
      float e = expf(sh_dots[row * KPAD_ + j] - mm);
      sh_dots[row * KPAD_ + j] = e;
      ssum += e;
    }
    sh_red[row * 8 + sub] = ssum;
    __syncthreads();
    if (sub == 0) {
      float ss = 0.f;
      for (int k2 = 0; k2 < 8; ++k2) ss += sh_red[row * 8 + k2];
      sh_rowsum[row] = ss;
    }
    __syncthreads();
    float inv = 1.0f / sh_rowsum[row];
    for (int j = sub; j < KPAD_; j += 8)
      sh_p[row * KPAD_ + j] = (f16)(sh_dots[row * KPAD_ + j] * inv);
  }
  __syncthreads();

  { // phase 2: out(16 x 64) = P(16 x 224) . q(224 x 64); wave -> 16 d-cols
    int prow = lane & 15;
    int nd = wave * 16 + (lane & 15);
    v8f c = v8f_zero();
    for (int ks = 0; ks < 7; ++ks) {
      int ka = ks * 32 + aks;
      v16h a = ld_frag_split(&sh_p[prow * KPAD_ + ka], &sh_p[prow * KPAD_ + ka + 16]);
      int kb = ks * 32 + bk0;
      v16h bf = ld_frag16(&sh_qT[nd * KPAD_ + kb]);
      c = wmma16(a, bf, c);
    }
    #pragma unroll
    for (int r = 0; r < 8; ++r) {
      int i = mtile * 16 + rowtop + r;
      if (i < S_) out16[((long)b * S_ + i) * C_ + h * D_ + nd] = (f16)c[r];
    }
  }
}

// ------------------------------------- mask + residual + LayerNorm -> f16
__global__ void resid_ln_k(const float* __restrict__ proj, const float* __restrict__ xa32,
                           const unsigned char* __restrict__ xm,
                           const float* __restrict__ g, const float* __restrict__ beta,
                           f16* __restrict__ out) {
  int b = blockIdx.x / L_;
  int l = blockIdx.x % L_;
  int tid = threadIdx.x;
  const float* pr = proj + ((long)b * L_ + l) * C_;
  const float* rx = xa32 + ((long)b * S_ + LT_ + l) * C_;
  float m = (xm[b * L_ + l] != 0) ? 1.0f : 0.0f;
  int c0 = tid, c1 = tid + 256;
  float v0 = pr[c0] * m + rx[c0];
  float v1 = pr[c1] * m + rx[c1];
  __shared__ float rbuf[256];
  rbuf[tid] = v0 + v1;
  __syncthreads();
  for (int s = 128; s > 0; s >>= 1) { if (tid < s) rbuf[tid] += rbuf[tid + s]; __syncthreads(); }
  float mean = rbuf[0] * (1.0f / (float)C_);
  __syncthreads();
  float d0 = v0 - mean, d1 = v1 - mean;
  rbuf[tid] = d0 * d0 + d1 * d1;
  __syncthreads();
  for (int s = 128; s > 0; s >>= 1) { if (tid < s) rbuf[tid] += rbuf[tid + s]; __syncthreads(); }
  float var = rbuf[0] * (1.0f / (float)C_);
  float rstd = rsqrtf(var + 1e-5f);
  f16* o = out + ((long)b * L_ + l) * C_;
  o[c0] = (f16)(d0 * rstd * g[c0] + beta[c0]);
  o[c1] = (f16)(d1 * rstd * g[c1] + beta[c1]);
}

// ------------------------------------------------------------------ launch
extern "C" void kernel_launch(void* const* d_in, const int* in_sizes, int n_in,
                              void* d_out, int out_size, void* d_ws, size_t ws_size,
                              hipStream_t stream) {
  const float* x            = (const float*)d_in[0];
  const unsigned char* xmsk = (const unsigned char*)d_in[1];
  const float* te1          = (const float*)d_in[2];
  const unsigned char* tmsk = (const unsigned char*)d_in[3];
  const float* dw_w  = (const float*)d_in[6];
  const float* dw_b  = (const float*)d_in[7];
  const float* pw_w  = (const float*)d_in[8];
  const float* pw_b  = (const float*)d_in[9];
  const float* wt    = (const float*)d_in[10];
  const float* bt    = (const float*)d_in[11];
  const float* rpe   = (const float*)d_in[12];
  const float* w_out = (const float*)d_in[13];
  const float* b_out = (const float*)d_in[14];
  const float* ln_g  = (const float*)d_in[15];
  const float* ln_b  = (const float*)d_in[16];
  const float* wp    = (const float*)d_in[17];
  const float* bp    = (const float*)d_in[18];
  float* out = (float*)d_out;
  (void)in_sizes; (void)n_in; (void)out_size; (void)ws_size;

  char* wsp = (char*)d_ws;
  auto carve = [&](size_t bytes) -> void* {
    void* p = (void*)wsp;
    wsp += (bytes + 255) & ~(size_t)255;
    return p;
  };
  f16*   pw_w16 = (f16*)carve((size_t)C_ * C_ * 2);
  f16*   wt16   = (f16*)carve((size_t)C_ * TD_ * 2);
  f16*   wout16 = (f16*)carve((size_t)C_ * C_ * 2);
  f16*   wp16   = (f16*)carve((size_t)C_ * C_ * 2);
  f16*   te116  = (f16*)carve((size_t)B_ * LT_ * TD_ * 2);
  f16*   hdw16  = (f16*)carve((size_t)B_ * L_ * C_ * 2);
  float* xa32   = (float*)carve((size_t)B_ * S_ * C_ * 4);
  f16*   xa16   = (f16*)carve((size_t)B_ * SPAD_ * C_ * 2);
  f16*   attn16 = (f16*)carve((size_t)B_ * S_ * C_ * 2);
  float* proj32 = (float*)carve((size_t)B_ * L_ * C_ * 4);
  f16*   h216   = (f16*)carve((size_t)B_ * L_ * C_ * 2);

  cvt_f32_to_f16_k<<<dim3(512), dim3(256), 0, stream>>>(pw_w,  pw_w16, C_ * C_);
  cvt_f32_to_f16_k<<<dim3(768), dim3(256), 0, stream>>>(wt,    wt16,   C_ * TD_);
  cvt_f32_to_f16_k<<<dim3(512), dim3(256), 0, stream>>>(w_out, wout16, C_ * C_);
  cvt_f32_to_f16_k<<<dim3(512), dim3(256), 0, stream>>>(wp,    wp16,   C_ * C_);
  cvt_f32_to_f16_k<<<dim3(768), dim3(256), 0, stream>>>(te1,   te116,  B_ * LT_ * TD_);

  dwconv_silu_k<<<dim3(L_ / 32, C_ / 8, B_), dim3(256), 0, stream>>>(x, dw_w, dw_b, hdw16);

  // cond = silu(te1 @ wt^T + bt) -> xa rows [0, Lt)
  gemm_wmma_k<<<dim3(C_ / 64, 1, B_), dim3(128), 0, stream>>>(
      te116, (long)LT_ * TD_, TD_, wt16, TD_, TD_, bt,
      xa32, (long)S_ * C_, xa16, (long)SPAD_ * C_, C_, LT_, 1);

  // pointwise: h @ pw_w^T + pw_b -> xa rows [Lt, S)
  gemm_wmma_k<<<dim3(C_ / 64, 3, B_), dim3(128), 0, stream>>>(
      hdw16, (long)L_ * C_, C_, pw_w16, C_, C_, pw_b,
      xa32 + LT_ * C_, (long)S_ * C_, xa16 + LT_ * C_, (long)SPAD_ * C_, C_, L_, 0);

  attn_k<<<dim3(13, H_, B_), dim3(128), 0, stream>>>(xa16, rpe, tmsk, xmsk, attn16);

  // out projection (only rows Lt.. are needed downstream)
  gemm_wmma_k<<<dim3(C_ / 64, 3, B_), dim3(128), 0, stream>>>(
      attn16 + LT_ * C_, (long)S_ * C_, C_, wout16, C_, C_, b_out,
      proj32, (long)L_ * C_, (f16*)nullptr, 0L, C_, L_, 0);

  resid_ln_k<<<dim3(B_ * L_), dim3(256), 0, stream>>>(proj32, xa32, xmsk, ln_g, ln_b, h216);

  // final: silu(h2 @ wp^T + bp) -> d_out (B, L, C)
  gemm_wmma_k<<<dim3(C_ / 64, 3, B_), dim3(128), 0, stream>>>(
      h216, (long)L_ * C_, C_, wp16, C_, C_, bp,
      out, (long)L_ * C_, (f16*)nullptr, 0L, C_, L_, 1);
}
